// ClassAttention_29557964931397
// MI455X (gfx1250) — matrix operand order repaired and checked by reading, not compiled, not run on hardware
//
#include <hip/hip_runtime.h>
#include <math.h>

// Problem constants (from reference)
#define B_   16
#define E_   256
#define NC_  8
#define Q_   40
#define HW_  4096
#define NH_  8
#define HD_  32
#define OF_  64

typedef __attribute__((ext_vector_type(16))) __bf16    v16bf;
typedef __attribute__((ext_vector_type(8)))  __bf16    v8bf;
typedef __attribute__((ext_vector_type(8)))  float     v8f;
typedef __attribute__((ext_vector_type(4)))  unsigned  v4u;
typedef __attribute__((ext_vector_type(8)))  unsigned  v8u;

// ---------- small helpers ----------
__device__ __forceinline__ unsigned short f2bf(float f) {
  unsigned u = __float_as_uint(f);
  unsigned r = u + 0x7FFFu + ((u >> 16) & 1u);   // round-to-nearest-even
  return (unsigned short)(r >> 16);
}

// Build a 16-element bf16 fragment from two 16B-aligned 8-element runs.
__device__ __forceinline__ v16bf frag_from2(const unsigned short* p0,
                                            const unsigned short* p1) {
  v4u lo = *(const v4u*)p0;
  v4u hi = *(const v4u*)p1;
  v8bf l = __builtin_bit_cast(v8bf, lo);
  v8bf h = __builtin_bit_cast(v8bf, hi);
  return __builtin_shufflevector(l, h, 0, 1, 2, 3, 4, 5, 6, 7,
                                 8, 9, 10, 11, 12, 13, 14, 15);
}
// Build a fragment from one contiguous 32B-aligned 16-element run.
__device__ __forceinline__ v16bf frag_from1(const unsigned short* p) {
  v8u r = *(const v8u*)p;
  return __builtin_bit_cast(v16bf, r);
}

__device__ __forceinline__ v8f wmma_bf16(v16bf a, v16bf b, v8f c) {
  return __builtin_amdgcn_wmma_f32_16x16x32_bf16(false, a, false, b, (short)0, c,
                                                 false, false);
}

// Async copy one 16-byte segment global -> LDS (ASYNCcnt-tracked).
__device__ __forceinline__ void async_g2l_b128(unsigned lds_off, const void* gptr) {
  unsigned long long g = (unsigned long long)(uintptr_t)gptr;
  asm volatile("global_load_async_to_lds_b128 %0, %1, off"
               :: "v"(lds_off), "v"(g) : "memory");
}
__device__ __forceinline__ void wait_async0() {
  asm volatile("s_wait_asynccnt 0x0" ::: "memory");
}

// blockDim.x == 256 reductions
__device__ __forceinline__ float block_reduce_sum(float v, float* sm) {
  int t = threadIdx.x;
  sm[t] = v; __syncthreads();
  for (int s = 128; s > 0; s >>= 1) { if (t < s) sm[t] += sm[t + s]; __syncthreads(); }
  float r = sm[0]; __syncthreads();
  return r;
}

// ---------- LayerNorm kernels ----------
__global__ __launch_bounds__(256)
void lnq_kernel(const float* __restrict__ X, const float* __restrict__ g,
                const float* __restrict__ be, unsigned short* __restrict__ out) {
  __shared__ float sm[256];
  const int row = blockIdx.x, t = threadIdx.x;
  float x = X[(size_t)row * E_ + t];
  float mean = block_reduce_sum(x, sm) * (1.0f / E_);
  float d = x - mean;
  float var = block_reduce_sum(d * d, sm) * (1.0f / E_);
  out[(size_t)row * E_ + t] = f2bf(d * rsqrtf(var + 1e-5f) * g[t] + be[t]);
}

__global__ __launch_bounds__(256)
void ln2_kernel(float* __restrict__ X, const float* __restrict__ g,
                const float* __restrict__ be) {
  __shared__ float sm[256];
  const int row = blockIdx.x, t = threadIdx.x;
  float x = X[(size_t)row * E_ + t];
  float mean = block_reduce_sum(x, sm) * (1.0f / E_);
  float d = x - mean;
  float var = block_reduce_sum(d * d, sm) * (1.0f / E_);
  X[(size_t)row * E_ + t] = d * rsqrtf(var + 1e-5f) * g[t] + be[t];
}

// ---------- pack concat(image,mask)^T -> bf16 [B,HW,288] (K zero-padded) ----------
__global__ __launch_bounds__(256)
void pack_kernel(const float* __restrict__ img, const float* __restrict__ msk,
                 unsigned short* __restrict__ xT) {
  size_t i = (size_t)blockIdx.x * 256 + threadIdx.x;   // over 16*4096*288
  int kc = (int)(i % 288);
  size_t r = i / 288;
  int hw = (int)(r & 4095), b = (int)(r >> 12);
  float v = 0.f;
  if (kc < 256)       v = img[((size_t)b * E_ + kc) * HW_ + hw];
  else if (kc < 264)  v = msk[((size_t)b * NC_ + (kc - 256)) * HW_ + hw];
  xT[i] = f2bf(v);
}

// ---------- generic WMMA GEMM: D = act(A_bf16[MxKpad] * Bw_f32[Kfull x 256] + bias) ----------
// modes: 0 f32 row-major, 1 bf16 row-major, 2 gelu+bf16 row-major,
//        3 bf16 scatter to kS fragment layout, 4 bf16 scatter to vS fragment layout
__global__ __launch_bounds__(128)
void gemm_wmma_kernel(const unsigned short* __restrict__ A, int lda,
                      const float* __restrict__ Bw, int Kfull, int Kpad,
                      const float* __restrict__ bias, int mode,
                      float* __restrict__ outF, unsigned short* __restrict__ outU) {
  __shared__ __align__(32) unsigned short lds_a[64 * 32];   // A block: 64 rows x 32 bf16
  __shared__ __align__(32) unsigned short btT[16 * 48];     // B tile transposed [n][k], padded
  const int tid = threadIdx.x;
  const int wave = tid >> 5, lane = tid & 31;
  const int m16 = lane & 15, half = lane >> 4;
  const int n0 = blockIdx.x * 16;
  const int row0 = blockIdx.y * 64;
  const unsigned ldsA0 = (unsigned)(uintptr_t)(void*)lds_a;

  v8f acc;
  const float bv = bias[n0 + m16];
  #pragma unroll
  for (int r = 0; r < 8; ++r) acc[r] = bv;

  const int nchunks = Kpad >> 5;
  for (int kc = 0; kc < nchunks; ++kc) {
    // 1) async-stage A block (64 x 32 bf16 = 4KB) into LDS: 256 x 16B segments
    {
      int s0 = tid, s1 = tid + 128;
      async_g2l_b128(ldsA0 + (unsigned)s0 * 16u,
                     A + (size_t)(row0 + (s0 >> 2)) * lda + kc * 32 + (s0 & 3) * 8);
      async_g2l_b128(ldsA0 + (unsigned)s1 * 16u,
                     A + (size_t)(row0 + (s1 >> 2)) * lda + kc * 32 + (s1 & 3) * 8);
    }
    // 2) cooperative f32 -> bf16 weight tile, transposed into LDS [n][k]
    {
      int i = tid * 4;
      int k = i >> 4, n = i & 15;
      int kg = kc * 32 + k;
      float4 w4;
      if (kg < Kfull) w4 = *(const float4*)(Bw + (size_t)kg * 256 + n0 + n);
      else            w4 = make_float4(0.f, 0.f, 0.f, 0.f);
      btT[(n + 0) * 48 + k] = f2bf(w4.x);
      btT[(n + 1) * 48 + k] = f2bf(w4.y);
      btT[(n + 2) * 48 + k] = f2bf(w4.z);
      btT[(n + 3) * 48 + k] = f2bf(w4.w);
    }
    wait_async0();
    __syncthreads();
    const unsigned short* ar = lds_a + (wave * 16 + m16) * 32;
    v16bf a  = frag_from2(ar + 8 * half, ar + 16 + 8 * half);
    v16bf bf = frag_from2(&btT[m16 * 48 + 8 * half], &btT[m16 * 48 + 16 + 8 * half]);
    acc = wmma_bf16(a, bf, acc);
    __syncthreads();
  }

  const int ncol = n0 + m16;
  #pragma unroll
  for (int r = 0; r < 8; ++r) {
    const int m = row0 + wave * 16 + r + 8 * half;
    float vl = acc[r];
    if (mode == 0) {
      outF[(size_t)m * 256 + ncol] = vl;
    } else if (mode == 1) {
      outU[(size_t)m * 256 + ncol] = f2bf(vl);
    } else if (mode == 2) {
      vl = 0.5f * vl * (1.0f + erff(vl * 0.70710678118654752f));  // exact gelu
      outU[(size_t)m * 256 + ncol] = f2bf(vl);
    } else if (mode == 3) {
      // kS[bn][hwTile][halfC][hw&15][c&15] : scores-kernel B-fragment order
      int b = m >> 12, hw = m & 4095;
      int nh = ncol >> 5, c = ncol & 31;
      size_t bn = (size_t)(b * NH_ + nh);
      outU[(((bn * 256 + (hw >> 4)) * 2 + (c >> 4)) * 16 + (hw & 15)) * 16 + (c & 15)] =
          f2bf(vl);
    } else {
      // vS[bn][kc][halfK][c][k&15] : pv-kernel B-fragment order
      int b = m >> 12, hw = m & 4095;
      int nh = ncol >> 5, c = ncol & 31;
      size_t bn = (size_t)(b * NH_ + nh);
      outU[(((bn * 128 + (hw >> 5)) * 2 + ((hw >> 4) & 1)) * 32 + c) * 16 + (hw & 15)] =
          f2bf(vl);
    }
  }
}

// ---------- depthwise 3x3 conv (SAME) -> bf16 [B*HW, E] ----------
__global__ __launch_bounds__(256)
void dwconv_kernel(const float* __restrict__ x2, const float* __restrict__ w,
                   const float* __restrict__ bias, unsigned short* __restrict__ xdw) {
  const int bhw = blockIdx.x;
  const int e = threadIdx.x;
  const int b = bhw >> 12, hw = bhw & 4095, h = hw >> 6, wx = hw & 63;
  float acc = bias[e];
  #pragma unroll
  for (int dy = -1; dy <= 1; ++dy)
    #pragma unroll
    for (int dx = -1; dx <= 1; ++dx) {
      int hh = h + dy, ww = wx + dx;
      if (hh >= 0 && hh < 64 && ww >= 0 && ww < 64)
        acc += x2[((size_t)(b * HW_ + hh * 64 + ww)) * E_ + e] *
               w[((dy + 1) * 3 + (dx + 1)) * E_ + e];
    }
  xdw[(size_t)bhw * E_ + e] = f2bf(acc);
}

// ---------- channel reductions for class bias ----------
__global__ __launch_bounds__(256)
void chanred_kernel(const float* __restrict__ X, int C, const float* __restrict__ w,
                    const float* __restrict__ b0, float* __restrict__ out) {
  const int b = blockIdx.y;
  const int hw = blockIdx.x * 256 + threadIdx.x;
  float acc = b0[0];
  for (int c = 0; c < C; ++c) acc += X[((size_t)b * C + c) * HW_ + hw] * w[c];
  out[(size_t)b * HW_ + hw] = acc;
}

__global__ __launch_bounds__(64)
void proj_kernel(const float* __restrict__ in, const float* __restrict__ w,
                 const float* __restrict__ bias, float* __restrict__ out) {
  const int b = blockIdx.x, of = threadIdx.x;
  float acc = bias[of];
  for (int k = 0; k < HW_; ++k) acc += in[(size_t)b * HW_ + k] * w[(size_t)k * OF_ + of];
  out[b * OF_ + of] = acc;
}

// ---------- attention scores (WMMA, fused class-bias + mask + scale) ----------
__global__ __launch_bounds__(32)
void scores_kernel(const unsigned short* __restrict__ qbf,
                   const unsigned short* __restrict__ kS,
                   const float* __restrict__ atti, const float* __restrict__ attm,
                   const float* __restrict__ cw, const float* __restrict__ cb,
                   const float* __restrict__ alpha, const float* __restrict__ beta,
                   const float* __restrict__ masks, float* __restrict__ scores) {
  const int bn = blockIdx.x;
  const int b = bn >> 3, nh = bn & 7;
  const int hwt = blockIdx.y;                 // 16-wide hw tile
  const int lane = threadIdx.x, m16 = lane & 15, half = lane >> 4;

  v16bf bf = frag_from1(kS + ((((size_t)bn * 256 + hwt) * 2 + half) * 16 + m16) * 16);

  const unsigned short* Abase = qbf + (size_t)b * Q_ * E_ + nh * HD_;
  v8f acc[3];
  #pragma unroll
  for (int mt = 0; mt < 3; ++mt) {
    #pragma unroll
    for (int r = 0; r < 8; ++r) acc[mt][r] = 0.f;
    const unsigned short* ar = Abase + (size_t)(mt * 16 + m16) * E_;
    v16bf a = frag_from1(ar + 16 * half);
    acc[mt] = wmma_bf16(a, bf, acc[mt]);
  }
  const int hw = hwt * 16 + m16;
  const float ai = atti[b * OF_ + (hw >> 6)];
  const float am = attm[b * OF_ + (hw & 63)];
  const float al = alpha[hw], be = beta[hw];
  const float inv = 0.17677669529663688f;  // 1/sqrt(32)
  #pragma unroll
  for (int mt = 0; mt < 3; ++mt)
    #pragma unroll
    for (int r = 0; r < 8; ++r) {
      int q = mt * 16 + r + 8 * half;
      if (q < Q_) {
        float cls = (ai * am * cw[q] + cb[q]) * al;
        float mk = masks[((size_t)(b * NC_ + q / 5)) * HW_ + hw];
        scores[((size_t)bn * Q_ + q) * HW_ + hw] = (cls + acc[mt][r] * mk * be) * inv;
      }
    }
}

// ---------- row softmax -> bf16 probs (48 padded rows per bn) ----------
__global__ __launch_bounds__(256)
void softmax_kernel(const float* __restrict__ scores, unsigned short* __restrict__ P) {
  const int row = blockIdx.x;
  const int q = row % 48, bn = row / 48;
  unsigned short* prow = P + (size_t)row * HW_;
  const int t = threadIdx.x;
  if (q >= Q_) {
    for (int i = t; i < HW_; i += 256) prow[i] = 0;
    return;
  }
  const float* srow = scores + ((size_t)bn * Q_ + q) * HW_;
  __shared__ float sm[256];
  float loc[16];
  float mx = -3.4e38f;
  #pragma unroll
  for (int k = 0; k < 16; ++k) { loc[k] = srow[t + k * 256]; mx = fmaxf(mx, loc[k]); }
  sm[t] = mx; __syncthreads();
  for (int s = 128; s > 0; s >>= 1) { if (t < s) sm[t] = fmaxf(sm[t], sm[t + s]); __syncthreads(); }
  mx = sm[0]; __syncthreads();
  float sum = 0.f;
  #pragma unroll
  for (int k = 0; k < 16; ++k) { loc[k] = __expf(loc[k] - mx); sum += loc[k]; }
  sum = block_reduce_sum(sum, sm);
  const float invs = 1.0f / sum;
  #pragma unroll
  for (int k = 0; k < 16; ++k) prow[t + k * 256] = f2bf(loc[k] * invs);
}

// ---------- out_h = P @ v (WMMA over K = 4096, fragment-swizzled v) ----------
__global__ __launch_bounds__(32)
void pv_kernel(const unsigned short* __restrict__ P, const unsigned short* __restrict__ vS,
               unsigned short* __restrict__ outh) {
  const int bn = blockIdx.x;
  const int mt = blockIdx.y >> 1, nt = blockIdx.y & 1;
  const int lane = threadIdx.x, m16 = lane & 15, half = lane >> 4;
  const unsigned short* Arow = P + ((size_t)bn * 48 + mt * 16 + m16) * HW_;
  v8f acc;
  #pragma unroll
  for (int r = 0; r < 8; ++r) acc[r] = 0.f;
  for (int kc = 0; kc < HW_ / 32; ++kc) {
    if (kc + 1 < HW_ / 32) __builtin_prefetch(Arow + (kc + 1) * 32, 0, 1);
    v16bf a = frag_from1(Arow + kc * 32 + 16 * half);
    v16bf bf = frag_from1(vS + ((((size_t)bn * 128 + kc) * 2 + half) * 32 + nt * 16 + m16) * 16);
    acc = wmma_bf16(a, bf, acc);
  }
  const int b = bn >> 3, nh = bn & 7;
  #pragma unroll
  for (int r = 0; r < 8; ++r) {
    int q = mt * 16 + r + 8 * half;
    if (q < Q_)
      outh[((size_t)(b * Q_ + q)) * E_ + nh * HD_ + nt * 16 + m16] = f2bf(acc[r]);
  }
}

// ---------- host orchestration ----------
extern "C" void kernel_launch(void* const* d_in, const int* in_sizes, int n_in,
                              void* d_out, int out_size, void* d_ws, size_t ws_size,
                              hipStream_t stream) {
  (void)in_sizes; (void)n_in; (void)out_size; (void)ws_size;
  const float* query = (const float*)d_in[0];
  const float* img   = (const float*)d_in[1];
  const float* msk   = (const float*)d_in[2];
  const float* masks = (const float*)d_in[3];
  const float* ln_g  = (const float*)d_in[4];
  const float* ln_b  = (const float*)d_in[5];
  const float* qw    = (const float*)d_in[6];
  const float* qb    = (const float*)d_in[7];
  const float* xp1_w = (const float*)d_in[8];
  const float* xp1_b = (const float*)d_in[9];
  const float* xp2_w = (const float*)d_in[10];
  const float* xp2_b = (const float*)d_in[11];
  const float* ln2_g = (const float*)d_in[12];
  const float* ln2_b = (const float*)d_in[13];
  const float* dw_w  = (const float*)d_in[14];
  const float* dw_b  = (const float*)d_in[15];
  const float* kw    = (const float*)d_in[16];
  const float* kb    = (const float*)d_in[17];
  const float* vw    = (const float*)d_in[18];
  const float* vbias = (const float*)d_in[19];
  const float* ic_w  = (const float*)d_in[20];
  const float* ic_b  = (const float*)d_in[21];
  const float* mc_w  = (const float*)d_in[22];
  const float* mc_b  = (const float*)d_in[23];
  const float* ipw   = (const float*)d_in[24];
  const float* ipb   = (const float*)d_in[25];
  const float* mpw   = (const float*)d_in[26];
  const float* mpb   = (const float*)d_in[27];
  const float* cw    = (const float*)d_in[28];
  const float* cb    = (const float*)d_in[29];
  const float* alpha = (const float*)d_in[30];
  const float* beta  = (const float*)d_in[31];
  const float* ow    = (const float*)d_in[32];
  const float* ob    = (const float*)d_in[33];
  float* out = (float*)d_out;

  char* ws = (char*)d_ws;
  auto up = [](size_t x) { return (x + 255) & ~(size_t)255; };
  size_t o = 0;
  unsigned short* qln  = (unsigned short*)(ws + o); o += up((size_t)640 * 256 * 2);
  unsigned short* qbf  = (unsigned short*)(ws + o); o += up((size_t)656 * 256 * 2); // +pad rows
  unsigned short* xdw  = (unsigned short*)(ws + o); o += up((size_t)65536 * 256 * 2);
  unsigned short* kS   = (unsigned short*)(ws + o); o += up((size_t)128 * 32 * 4096 * 2);
  unsigned short* vS   = (unsigned short*)(ws + o); o += up((size_t)128 * 4096 * 32 * 2);
  unsigned short* outh = (unsigned short*)(ws + o); o += up((size_t)640 * 256 * 2);
  float* ai0  = (float*)(ws + o); o += up((size_t)16 * 4096 * 4);
  float* am0  = (float*)(ws + o); o += up((size_t)16 * 4096 * 4);
  float* atti = (float*)(ws + o); o += up((size_t)16 * 64 * 4);
  float* attm = (float*)(ws + o); o += up((size_t)16 * 64 * 4);
  // region A: xcatT + x1 early, reused by softmax probs P later
  size_t regA = o;
  unsigned short* xcatT = (unsigned short*)(ws + regA);
  unsigned short* x1    = (unsigned short*)(ws + regA + up((size_t)65536 * 288 * 2));
  unsigned short* P     = (unsigned short*)(ws + regA);
  size_t szA1 = up((size_t)65536 * 288 * 2) + up((size_t)65536 * 256 * 2);
  size_t szA2 = up((size_t)128 * 48 * 4096 * 2);
  size_t regB = regA + (szA1 > szA2 ? szA1 : szA2);
  // region B: x2 (f32) early, reused by scores (f32) later
  float* x2     = (float*)(ws + regB);
  float* scores = (float*)(ws + regB);

  // 1) q branch: LN -> bf16, then q = LN @ qw + qb (bf16)
  lnq_kernel<<<640, 256, 0, stream>>>(query, ln_g, ln_b, qln);
  gemm_wmma_kernel<<<dim3(16, 10), 128, 0, stream>>>(qln, 256, qw, 256, 256, qb, 1, nullptr, qbf);

  // 2) x branch
  pack_kernel<<<73728, 256, 0, stream>>>(img, msk, xcatT);
  gemm_wmma_kernel<<<dim3(16, 1024), 128, 0, stream>>>(xcatT, 288, xp1_w, 264, 288, xp1_b, 2, nullptr, x1);
  gemm_wmma_kernel<<<dim3(16, 1024), 128, 0, stream>>>(x1, 256, xp2_w, 256, 256, xp2_b, 0, x2, nullptr);
  ln2_kernel<<<65536, 256, 0, stream>>>(x2, ln2_g, ln2_b);
  dwconv_kernel<<<65536, 256, 0, stream>>>(x2, dw_w, dw_b, xdw);
  gemm_wmma_kernel<<<dim3(16, 1024), 128, 0, stream>>>(xdw, 256, kw, 256, 256, kb, 3, nullptr, kS);
  gemm_wmma_kernel<<<dim3(16, 1024), 128, 0, stream>>>(xdw, 256, vw, 256, 256, vbias, 4, nullptr, vS);

  // 3) class-attention additive bias path
  chanred_kernel<<<dim3(16, 16), 256, 0, stream>>>(img, 256, ic_w, ic_b, ai0);
  chanred_kernel<<<dim3(16, 16), 256, 0, stream>>>(msk, 8, mc_w, mc_b, am0);
  proj_kernel<<<16, 64, 0, stream>>>(ai0, ipw, ipb, atti);
  proj_kernel<<<16, 64, 0, stream>>>(am0, mpw, mpb, attm);

  // 4) attention
  scores_kernel<<<dim3(128, 256), 32, 0, stream>>>(qbf, kS, atti, attm, cw, cb, alpha, beta, masks, scores);
  softmax_kernel<<<128 * 48, 256, 0, stream>>>(scores, P);
  pv_kernel<<<dim3(128, 6), 32, 0, stream>>>(P, vS, outh);

  // 5) output projection -> d_out (f32)
  gemm_wmma_kernel<<<dim3(16, 10), 128, 0, stream>>>(outh, 256, ow, 256, 256, ob, 0, out, nullptr);
}
